// ChildSumTreeLSTM_10900626997596
// MI455X (gfx1250) — matrix operand order repaired and checked by reading, not compiled, not run on hardware
//
#include <hip/hip_runtime.h>
#include <hip/hip_bf16.h>
#include <stdint.h>

// ---------------------------------------------------------------------------
// Child-Sum TreeLSTM on MI455X (gfx1250, wave32, WMMA).
//   B=64, N=512, D=300.  K padded to 320, N-cols padded to 304 (WMMA 16x16x32).
// Weights are pre-swizzled into the exact WMMA B-fragment word layout so the
// GEMM inner loops are pure b128 loads + v_wmma.
// ---------------------------------------------------------------------------
#define BDIM 64
#define NSEQ 512
#define DDIM 300
#define KPAD 320
#define NPAD 304
#define NKCHUNK (KPAD / 32)      // 10 WMMA K-steps of 32
#define NTILES  (NPAD / 16)      // 19 column tiles
#define REC_WGS NTILES           // one WG per e-tile
#define WHB_WG_WORDS (4 * NKCHUNK * 8 * 32)   // 10240 words = 40KB per e-tile

typedef __attribute__((ext_vector_type(16))) __bf16 v16bf;
typedef __attribute__((ext_vector_type(8)))  float  v8f;

union Frag {
    v16bf        v;
    unsigned int u[8];
    unsigned short s[16];
    uint4        q[2];
};

__device__ __forceinline__ v8f vzero() {
    v8f z = {0.f, 0.f, 0.f, 0.f, 0.f, 0.f, 0.f, 0.f};
    return z;
}
__device__ __forceinline__ unsigned short f2bf(float f) {
    unsigned int u = __float_as_uint(f);
    u += 0x7fffu + ((u >> 16) & 1u);     // round-to-nearest-even
    return (unsigned short)(u >> 16);
}
__device__ __forceinline__ float bf2f(unsigned short s) {
    return __uint_as_float(((unsigned int)s) << 16);
}
__device__ __forceinline__ float sigmoidf_(float x) {
    return 1.0f / (1.0f + __expf(-x));
}

// ---------------- workspace layout (bytes, all 256-aligned) ----------------
// HSUM  : f32 [64][513][320]    master child-sum accumulator (pad cols stay 0)
// FC    : f32 [64][513][320]    master f*c accumulator
// HSUMB : bf16 [64][513][320]   bf16 mirror of HSUM (WMMA A operand)
// ST    : bf16 [64][320]        per-step st staging (pad cols stay 0)
// CNT   : u32 barrier counter
// ABF   : bf16 [32768][320]     embs, K-padded
// W5P   : u32 [5][19][10][32][8]    phase-1 B fragments (two bf16/word)
// WHBP  : u32 [19][4][10][8][32]    recurrent B fragments, == LDS image per tile
// GX    : bf16 [4][512][64][304]    (k,t,b,e): each step reads contiguously
// GATE  : bf16 [32768][304]     tanh(embs@Wt+bt)
static const size_t OFF_HSUM  = 0;
static const size_t SZ_HSUM   = (size_t)BDIM * (NSEQ + 1) * KPAD * 4;     // 42,024,960
static const size_t OFF_FC    = OFF_HSUM + SZ_HSUM;
static const size_t OFF_HSUMB = OFF_FC + SZ_HSUM;
static const size_t SZ_HSUMB  = (size_t)BDIM * (NSEQ + 1) * KPAD * 2;     // 21,012,480
static const size_t OFF_ST    = OFF_HSUMB + SZ_HSUMB;
static const size_t SZ_ST     = (size_t)BDIM * KPAD * 2;                  // 40,960
static const size_t OFF_CNT   = OFF_ST + SZ_ST;
static const size_t OFF_ABF   = OFF_CNT + 256;
static const size_t SZ_ABF    = (size_t)BDIM * NSEQ * KPAD * 2;           // 20,971,520
static const size_t OFF_W5P   = OFF_ABF + SZ_ABF;
static const size_t W5P_WORDS = (size_t)5 * NTILES * NKCHUNK * 32 * 8;    // 243,200
static const size_t OFF_WHBP  = OFF_W5P + W5P_WORDS * 4;
static const size_t WHBP_WORDS= (size_t)NTILES * WHB_WG_WORDS;            // 194,560
static const size_t OFF_GX    = OFF_WHBP + WHBP_WORDS * 4;
static const size_t SZ_GX     = (size_t)4 * NSEQ * BDIM * NPAD * 2;       // 79,691,776
static const size_t OFF_GATE  = OFF_GX + SZ_GX;
static const size_t ZERO_WORDS = OFF_ABF / 4;   // HSUM+FC+HSUMB+ST+CNT zeroed each call

// ---------------------------------------------------------------------------
__global__ void k_init(unsigned int* p, size_t nwords) {
    size_t stride = (size_t)gridDim.x * blockDim.x;
    for (size_t i = (size_t)blockIdx.x * blockDim.x + threadIdx.x; i < nwords; i += stride)
        p[i] = 0u;
}

// embs (f32, 32768x300) -> bf16 (32768x320, K zero-padded)
__global__ void k_convert_embs(const float* __restrict__ embs, unsigned short* __restrict__ abf) {
    size_t total = (size_t)BDIM * NSEQ * KPAD;
    size_t stride = (size_t)gridDim.x * blockDim.x;
    for (size_t i = (size_t)blockIdx.x * blockDim.x + threadIdx.x; i < total; i += stride) {
        size_t r = i / KPAD, c = i % KPAD;
        abf[i] = f2bf(c < DDIM ? embs[r * DDIM + c] : 0.f);
    }
}

// ---------------------------------------------------------------------------
// Weights -> bf16, zero-padded, PRE-SWIZZLED into WMMA B-fragment word layout.
// One u32 word = bf16 pair for rows (kb+2(r&3), kb+2(r&3)+1) at column n, where for
// lane L: n = ntile*16 + (L&15), kb = kchunk*32 + (r&4 ? 16 : 0) + (L&16 ? 8 : 0).
//   W5P : word[((z*19+nt)*10+kk)*32 + L)*8 + r]    (per-lane contiguous -> b128)
//   WHBP: word[nt*10240 + ((k*10+kk)*8 + r)*32 + L] (== recurrent LDS image)
// ---------------------------------------------------------------------------
__global__ void k_convert_w(const float* __restrict__ Wx, const float* __restrict__ Wt,
                            const float* __restrict__ Wh,
                            unsigned int* __restrict__ W5P, unsigned int* __restrict__ WHBP) {
    const size_t total5 = W5P_WORDS;
    const size_t total4 = WHBP_WORDS;
    size_t stride = (size_t)gridDim.x * blockDim.x;
    for (size_t i = (size_t)blockIdx.x * blockDim.x + threadIdx.x; i < total5 + total4; i += stride) {
        if (i < total5) {
            size_t idx = i;
            const int r  = idx & 7;        idx >>= 3;
            const int L  = idx & 31;       idx >>= 5;
            const int kk = idx % NKCHUNK;  idx /= NKCHUNK;
            const int nt = idx % NTILES;
            const int z  = idx / NTILES;
            const int n  = nt * 16 + (L & 15);
            const int row = kk * 32 + ((r & 4) ? 16 : 0) + ((L & 16) ? 8 : 0) + 2 * (r & 3);
            const float* src = (z < 4) ? (Wx + (size_t)z * DDIM * DDIM) : Wt;
            const bool nv = (n < DDIM);
            const unsigned int lo = f2bf((nv && row     < DDIM) ? src[(size_t)row       * DDIM + n] : 0.f);
            const unsigned int hi = f2bf((nv && row + 1 < DDIM) ? src[(size_t)(row + 1) * DDIM + n] : 0.f);
            W5P[i] = lo | (hi << 16);
        } else {
            size_t j = i - total5;
            const int L  = j & 31;         j >>= 5;
            const int r  = j & 7;          j >>= 3;
            const int kk = j % NKCHUNK;    j /= NKCHUNK;
            const int k  = j % 4;
            const int nt = j / 4;
            const int n  = nt * 16 + (L & 15);
            const int row = kk * 32 + ((r & 4) ? 16 : 0) + ((L & 16) ? 8 : 0) + 2 * (r & 3);
            const float* src = Wh + (size_t)k * DDIM * DDIM;
            const bool nv = (n < DDIM);
            const unsigned int lo = f2bf((nv && row     < DDIM) ? src[(size_t)row       * DDIM + n] : 0.f);
            const unsigned int hi = f2bf((nv && row + 1 < DDIM) ? src[(size_t)(row + 1) * DDIM + n] : 0.f);
            WHBP[i - total5] = lo | (hi << 16);
        }
    }
}

// ---------------------------------------------------------------------------
// Phase-1 GEMM: gx[k]=embs@Wx[k]+bx[k] (z<4), gate=tanh(embs@Wt+bt) (z==4).
// grid (256, 19, 5), block 256 = 8 waves; inner loop = 4x b128 + 1x wmma.
// ---------------------------------------------------------------------------
__global__ void __launch_bounds__(256) k_gemm(
        const unsigned short* __restrict__ A, const unsigned int* __restrict__ W5P,
        const float* __restrict__ bx, const float* __restrict__ bt,
        unsigned short* __restrict__ GX, unsigned short* __restrict__ GATE) {
    const int lane = threadIdx.x & 31;
    const int wave = threadIdx.x >> 5;
    const int row0 = (blockIdx.x * 8 + wave) * 16;       // M tile (b*512+t rows)
    const int nt   = blockIdx.y;                         // N tile
    const int z    = blockIdx.z;                         // which matrix
    const int m    = lane & 15;
    const int hi   = (lane & 16) ? 1 : 0;
    const unsigned short* arow = A + (size_t)(row0 + m) * KPAD;
    const unsigned int*   bbase = W5P + ((((size_t)z * NTILES + nt) * NKCHUNK) * 32 + lane) * 8;

    v8f acc = vzero();
    for (int kk = 0; kk < NKCHUNK; ++kk) {
        Frag a, b;
        const int s1 = kk * 32 + hi * 8;     // A lanes 0-15: K={0..7,16..23}; 16-31: +8
        a.q[0] = *(const uint4*)(arow + s1);
        a.q[1] = *(const uint4*)(arow + s1 + 16);
        const uint4* bp = (const uint4*)(bbase + (size_t)kk * 32 * 8);
        b.q[0] = bp[0];
        b.q[1] = bp[1];
        acc = __builtin_amdgcn_wmma_f32_16x16x32_bf16(false, a.v, false, b.v,
                                                      (short)0, acc, false, false);
    }
    const int e  = nt * 16 + (lane & 15);
    const int eb = (e < DDIM) ? e : 0;
    float bias = (z < 4) ? bx[z * DDIM + eb] : bt[eb];
    if (e >= DDIM) bias = 0.f;
#pragma unroll
    for (int r = 0; r < 8; ++r) {                        // C: M = r + 8*hi, N = lane&15
        const int mrow = row0 + r + hi * 8;
        const int bidx = mrow >> 9;
        const int t    = mrow & 511;
        const float v  = acc[r] + bias;
        if (z == 4)
            GATE[(size_t)mrow * NPAD + e] = f2bf(tanhf(v));
        else
            GX[(((size_t)z * NSEQ + t) * BDIM + bidx) * NPAD + e] = f2bf(v);
    }
}

// ---------------------------------------------------------------------------
// Device-scope grid barrier (19 resident WGs; counter monotonically increases)
// ---------------------------------------------------------------------------
__device__ __forceinline__ void grid_barrier(unsigned int* cnt, unsigned int target) {
    __threadfence();
    __syncthreads();
    if (threadIdx.x == 0) {
        __hip_atomic_fetch_add(cnt, 1u, __ATOMIC_RELEASE, __HIP_MEMORY_SCOPE_AGENT);
        while (__hip_atomic_load(cnt, __ATOMIC_ACQUIRE, __HIP_MEMORY_SCOPE_AGENT) < target)
            __builtin_amdgcn_s_sleep(2);
    }
    __syncthreads();
    __threadfence();
}

// ---------------------------------------------------------------------------
// Persistent recurrent kernel: 19 WGs x 128 threads (4 waves).
// WG owns a 16-col e-tile; its pre-swizzled Wh slice (40 KB) is memcpy'd into
// LDS once and reused for all 512 steps. Wave owns a 16-row b-tile; per step:
// 3 WMMA chains (i,o,u) with bf16 HSUMB fragments, elementwise cell, barrier,
// 1 WMMA chain (f = st@Wh3), parent scatter-add (f32 master + bf16 mirror),
// barrier.
// ---------------------------------------------------------------------------
__global__ void __launch_bounds__(128) k_recurrent(
        const float* __restrict__ embs, const float* __restrict__ bh,
        const int* __restrict__ parent,
        const unsigned int* __restrict__ WHBP,
        const unsigned short* __restrict__ GX, const unsigned short* __restrict__ GATE,
        float* __restrict__ HSUM, float* __restrict__ FC,
        unsigned short* __restrict__ HSUMB, unsigned short* __restrict__ ST,
        unsigned int* __restrict__ CNT, float* __restrict__ out) {
    __shared__ unsigned int ldsW[WHB_WG_WORDS];          // 40 KB of 320 KB LDS

    const int lane = threadIdx.x & 31;
    const int wave = threadIdx.x >> 5;
    const int nt   = blockIdx.x;
    const int n0   = nt * 16;                            // e-tile
    const int b0   = wave * 16;                          // b-tile

    // Coalesced copy: WHBP image for this tile is already in LDS layout.
    {
        const unsigned int* src = WHBP + (size_t)nt * WHB_WG_WORDS;
        for (int i = threadIdx.x; i < WHB_WG_WORDS; i += blockDim.x)
            ldsW[i] = src[i];
    }
    __syncthreads();

    const int  m      = lane & 15;
    const int  hi     = (lane & 16) ? 1 : 0;
    const int  e      = n0 + (lane & 15);
    const bool evalid = (e < DDIM);
    const int  eb     = evalid ? e : 0;
    const float bh0 = bh[0 * DDIM + eb], bh1 = bh[1 * DDIM + eb];
    const float bh2 = bh[2 * DDIM + eb], bh3 = bh[3 * DDIM + eb];

    unsigned int gen = 0;
    const unsigned int nwg = gridDim.x;

    for (int t = 0; t < NSEQ; ++t) {
        // ---------- phase 1: i,o,u gates + cell + st ----------
        v8f acc0 = vzero(), acc1 = vzero(), acc2 = vzero();
        const unsigned short* hrow = HSUMB + ((size_t)(b0 + m) * (NSEQ + 1) + t) * KPAD;
        for (int kk = 0; kk < NKCHUNK; ++kk) {
            Frag a;
            const int s1 = kk * 32 + hi * 8;
            a.q[0] = *(const uint4*)(hrow + s1);
            a.q[1] = *(const uint4*)(hrow + s1 + 16);
            Frag w0, w1, w2;
#pragma unroll
            for (int r = 0; r < 8; ++r) {
                w0.u[r] = ldsW[((0 * NKCHUNK + kk) * 8 + r) * 32 + lane];
                w1.u[r] = ldsW[((1 * NKCHUNK + kk) * 8 + r) * 32 + lane];
                w2.u[r] = ldsW[((2 * NKCHUNK + kk) * 8 + r) * 32 + lane];
            }
            acc0 = __builtin_amdgcn_wmma_f32_16x16x32_bf16(false, a.v, false, w0.v, (short)0, acc0, false, false);
            acc1 = __builtin_amdgcn_wmma_f32_16x16x32_bf16(false, a.v, false, w1.v, (short)0, acc1, false, false);
            acc2 = __builtin_amdgcn_wmma_f32_16x16x32_bf16(false, a.v, false, w2.v, (short)0, acc2, false, false);
        }
        float stv[8];
#pragma unroll
        for (int r = 0; r < 8; ++r) {
            const int bl = b0 + r + hi * 8;
            const size_t rowBT = (size_t)bl * NSEQ + t;
            const float gx0 = bf2f(GX[((0 * (size_t)NSEQ + t) * BDIM + bl) * NPAD + e]);
            const float gx1 = bf2f(GX[((1 * (size_t)NSEQ + t) * BDIM + bl) * NPAD + e]);
            const float gx2 = bf2f(GX[((2 * (size_t)NSEQ + t) * BDIM + bl) * NPAD + e]);
            const float fc  = FC[((size_t)bl * (NSEQ + 1) + t) * KPAD + e];
            const float iv = sigmoidf_(gx0 + acc0[r] + bh0);
            const float ov = sigmoidf_(gx1 + acc1[r] + bh1);
            const float uv = tanhf(gx2 + acc2[r] + bh2);
            const float c  = iv * uv + fc;
            const float h  = ov * tanhf(c);
            const float x  = embs[rowBT * DDIM + eb];
            const float g  = bf2f(GATE[rowBT * NPAD + e]);
            const float st = x * (1.f - g) + h * g;
            stv[r] = st;
            if (evalid) {
                out[rowBT * DDIM + e] = st;
                ST[(size_t)bl * KPAD + e] = f2bf(st);    // pad cols stay zero
            }
        }
        grid_barrier(CNT, ++gen * nwg);                  // all of st visible

        // ---------- phase 2: f = sigmoid(gx3[parent] + st@Wh3 + bh3); scatter ----------
        v8f acc3 = vzero();
        const unsigned short* strow = ST + (size_t)(b0 + m) * KPAD;
        for (int kk = 0; kk < NKCHUNK; ++kk) {
            Frag a, w3;
            const int s1 = kk * 32 + hi * 8;
            a.q[0] = *(const uint4*)(strow + s1);
            a.q[1] = *(const uint4*)(strow + s1 + 16);
#pragma unroll
            for (int r = 0; r < 8; ++r)
                w3.u[r] = ldsW[((3 * NKCHUNK + kk) * 8 + r) * 32 + lane];
            acc3 = __builtin_amdgcn_wmma_f32_16x16x32_bf16(false, a.v, false, w3.v, (short)0, acc3, false, false);
        }
#pragma unroll
        for (int r = 0; r < 8; ++r) {
            const int bl = b0 + r + hi * 8;
            const int p  = parent[bl * NSEQ + t];        // p in [t+1, 512]
            const float gx3 = (p < NSEQ)
                ? bf2f(GX[((3 * (size_t)NSEQ + p) * BDIM + bl) * NPAD + e]) : 0.f;
            const float f = sigmoidf_(gx3 + acc3[r] + bh3);
            if (evalid) {
                const size_t hidx = ((size_t)bl * (NSEQ + 1) + p) * KPAD + e;
                const float nh = HSUM[hidx] + stv[r];    // unique (b,e) owner: race-free
                HSUM[hidx]  = nh;
                HSUMB[hidx] = f2bf(nh);                  // bf16 mirror for WMMA A
                FC[hidx]   += f * stv[r];
            }
        }
        grid_barrier(CNT, ++gen * nwg);                  // updates visible before t+1
    }
}

// ---------------------------------------------------------------------------
extern "C" void kernel_launch(void* const* d_in, const int* in_sizes, int n_in,
                              void* d_out, int out_size, void* d_ws, size_t ws_size,
                              hipStream_t stream) {
    const float* embs  = (const float*)d_in[0];
    const float* Wx    = (const float*)d_in[1];
    const float* bx    = (const float*)d_in[2];
    const float* Wh    = (const float*)d_in[3];
    const float* bh    = (const float*)d_in[4];
    const float* Wt    = (const float*)d_in[5];
    const float* bt    = (const float*)d_in[6];
    const int*   parent= (const int*)d_in[7];
    float* out = (float*)d_out;

    char* ws = (char*)d_ws;   // needs ~220 MiB (see layout above)
    float*          HSUM  = (float*)(ws + OFF_HSUM);
    float*          FCp   = (float*)(ws + OFF_FC);
    unsigned short* HSUMB = (unsigned short*)(ws + OFF_HSUMB);
    unsigned short* ST    = (unsigned short*)(ws + OFF_ST);
    unsigned int*   CNT   = (unsigned int*)(ws + OFF_CNT);
    unsigned short* ABF   = (unsigned short*)(ws + OFF_ABF);
    unsigned int*   W5P   = (unsigned int*)(ws + OFF_W5P);
    unsigned int*   WHBP  = (unsigned int*)(ws + OFF_WHBP);
    unsigned short* GX    = (unsigned short*)(ws + OFF_GX);
    unsigned short* GATE  = (unsigned short*)(ws + OFF_GATE);

    k_init<<<2048, 256, 0, stream>>>((unsigned int*)ws, ZERO_WORDS);
    k_convert_embs<<<1024, 256, 0, stream>>>(embs, ABF);
    k_convert_w<<<512, 256, 0, stream>>>(Wx, Wt, Wh, W5P, WHBP);
    k_gemm<<<dim3(256, NTILES, 5), 256, 0, stream>>>(ABF, W5P, bx, bt, GX, GATE);
    k_recurrent<<<REC_WGS, 128, 0, stream>>>(embs, bh, parent, WHBP, GX, GATE,
                                             HSUM, FCp, HSUMB, ST, CNT, out);
    (void)in_sizes; (void)n_in; (void)out_size; (void)ws_size;
}
